// GReFELModel_33397665693867
// MI455X (gfx1250) — matrix operand order repaired and checked by reading, not compiled, not run on hardware
//
#include <hip/hip_runtime.h>
#include <hip/hip_bf16.h>
#include <cstdint>

typedef __attribute__((ext_vector_type(16))) _Float16 v16h;
typedef __attribute__((ext_vector_type(8)))  _Float16 v8h;
typedef __attribute__((ext_vector_type(8)))  float    v8f;

#define C_DIM   512
#define MLP_DIM 2048
#define NHEAD   8
#define HDIM    64
#define BATCH   16
#define DEPTH_N 6

#define TILE_M 128
#define TILE_N 128
#define TILE_K 32

// ------------------------------------------------------------------
// f32 -> f16 conversion (grid-stride)
// ------------------------------------------------------------------
__global__ void cvt_f32_f16_kernel(const float* __restrict__ s,
                                   _Float16* __restrict__ d, size_t n) {
  size_t i = (size_t)blockIdx.x * blockDim.x + threadIdx.x;
  size_t stride = (size_t)gridDim.x * blockDim.x;
  for (; i < n; i += stride) d[i] = (_Float16)s[i];
}

// f32 (Kreal x N) -> f16 (Kpad x N), zero-padding extra rows
__global__ void cvt_pad_rows_kernel(const float* __restrict__ s,
                                    _Float16* __restrict__ d,
                                    int kreal, int kpad, int n) {
  size_t total = (size_t)kpad * n;
  size_t i = (size_t)blockIdx.x * blockDim.x + threadIdx.x;
  size_t stride = (size_t)gridDim.x * blockDim.x;
  for (; i < total; i += stride) {
    int col = (int)(i % n);
    int row = (int)(i / n);
    d[i] = (row < kreal) ? (_Float16)s[(size_t)row * n + col] : (_Float16)0.f;
  }
}

// ------------------------------------------------------------------
// Patchify: images (B,3,224,224) f32 -> Xp (B*n, Kpad) f16, zero pad K
// row = b*n + (ih*nside+iw); col = ch*p*p + py*p + px
// ------------------------------------------------------------------
__global__ void patchify_kernel(const float* __restrict__ img,
                                _Float16* __restrict__ xp,
                                int p, int nside, int kreal, int kpad) {
  int npatch = nside * nside;
  size_t total = (size_t)BATCH * npatch * kpad;
  size_t i = (size_t)blockIdx.x * blockDim.x + threadIdx.x;
  size_t stride = (size_t)gridDim.x * blockDim.x;
  for (; i < total; i += stride) {
    int col = (int)(i % kpad);
    size_t row = i / kpad;
    if (col >= kreal) { xp[i] = (_Float16)0.f; continue; }
    int patch = (int)(row % npatch);
    int b = (int)(row / npatch);
    int ch = col / (p * p);
    int r = col % (p * p);
    int py = r / p, px = r % p;
    int ih = patch / nside, iw = patch % nside;
    xp[i] = (_Float16)img[(((size_t)b * 3 + ch) * 224 + (size_t)ih * p + py) * 224
                          + (size_t)iw * p + px];
  }
}

// ------------------------------------------------------------------
// CDNA5 async global->LDS copy of 16 bytes per lane (ASYNCcnt-tracked)
// ------------------------------------------------------------------
__device__ __forceinline__ void async_b128(const _Float16* g, _Float16* l) {
  unsigned lds = (unsigned)(uintptr_t)l;  // flat->LDS: addr[31:0]
  asm volatile("global_load_async_to_lds_b128 %0, %1, off"
               :: "v"(lds), "v"(g) : "memory");
}

// ------------------------------------------------------------------
// WMMA GEMM: out = act(A(MxK) @ B(KxN) + bias + pos) + resid
// A,B f16 row-major. K % 32 == 0, N % 128 == 0. M guarded.
// act: 0=none, 1=exact GELU, 2=ReLU. pos added as pos[(row%posN)*N+col].
// Block tile 128x128, 8 waves (4x2), wave tile 32x64 = 2x4 WMMA tiles.
// Double-buffered LDS filled via global_load_async_to_lds_b128.
// ------------------------------------------------------------------
__global__ __launch_bounds__(256) void gemm_wmma_kernel(
    const _Float16* __restrict__ A, const _Float16* __restrict__ B,
    const float* __restrict__ bias, const float* __restrict__ pos,
    const _Float16* __restrict__ resid,
    _Float16* __restrict__ outh, float* __restrict__ outf,
    int M, int N, int K, int posN, int act) {
  __shared__ _Float16 sA[2][TILE_M][TILE_K];  // 2 x 8 KB
  __shared__ _Float16 sB[2][TILE_K][TILE_N];  // 2 x 8 KB

  const int tid  = threadIdx.x;
  const int lane = tid & 31;
  const int wave = tid >> 5;        // 0..7
  const int wm   = wave >> 1;       // 0..3  (32-row strip)
  const int wn   = wave & 1;        // 0..1  (64-col strip)
  const int bm0  = blockIdx.x * TILE_M;
  const int bn0  = blockIdx.y * TILE_N;

  v8f acc[2][4];
#pragma unroll
  for (int tm = 0; tm < 2; ++tm)
#pragma unroll
    for (int tn = 0; tn < 4; ++tn)
#pragma unroll
      for (int j = 0; j < 8; ++j) acc[tm][tn][j] = 0.f;

  // staging coordinates: 32 bytes (2 x b128) per thread for each of A/B
  const int arow = tid >> 1;        // 0..127
  const int aseg = (tid & 1) * 16;  // halves
  const int brow = tid >> 3;        // 0..31
  const int bseg = (tid & 7) * 16;  // halves
  const int aRowG = (bm0 + arow < M) ? (bm0 + arow) : (M - 1);  // clamp edge

  // prologue: stage k0 = 0 into buffer 0
  {
    const _Float16* ga = A + (size_t)aRowG * K + aseg;
    async_b128(ga,     &sA[0][arow][aseg]);
    async_b128(ga + 8, &sA[0][arow][aseg + 8]);
    const _Float16* gb = B + (size_t)brow * N + bn0 + bseg;
    async_b128(gb,     &sB[0][brow][bseg]);
    async_b128(gb + 8, &sB[0][brow][bseg + 8]);
  }

  for (int k0 = 0; k0 < K; k0 += TILE_K) {
    const int buf = (k0 >> 5) & 1;
    if (k0 + TILE_K < K) {  // stage next tile into the other buffer
      const int nb = buf ^ 1;
      const _Float16* ga = A + (size_t)aRowG * K + (k0 + TILE_K) + aseg;
      async_b128(ga,     &sA[nb][arow][aseg]);
      async_b128(ga + 8, &sA[nb][arow][aseg + 8]);
      const _Float16* gb = B + (size_t)(k0 + TILE_K + brow) * N + bn0 + bseg;
      async_b128(gb,     &sB[nb][brow][bseg]);
      async_b128(gb + 8, &sB[nb][brow][bseg + 8]);
      asm volatile("s_wait_asynccnt 0x4" ::: "memory");  // current stage done
    } else {
      asm volatile("s_wait_asynccnt 0x0" ::: "memory");
    }
    __syncthreads();  // all waves' async fills of `buf` visible

    // fragments per documented CDNA5 layouts
    const int kb = (lane >> 4) << 3;   // 0 or 8
    v16h afrag[2], bfrag[4];
#pragma unroll
    for (int tm = 0; tm < 2; ++tm) {
      const _Float16* ap = &sA[buf][wm * 32 + tm * 16 + (lane & 15)][0];
      v8h lo = *(const v8h*)(ap + kb);
      v8h hi = *(const v8h*)(ap + kb + 16);
#pragma unroll
      for (int j = 0; j < 8; ++j) { afrag[tm][j] = lo[j]; afrag[tm][j + 8] = hi[j]; }
    }
#pragma unroll
    for (int tn = 0; tn < 4; ++tn)
#pragma unroll
      for (int j = 0; j < 8; ++j) {
        bfrag[tn][j]     = sB[buf][kb + j][wn * 64 + tn * 16 + (lane & 15)];
        bfrag[tn][j + 8] = sB[buf][kb + 16 + j][wn * 64 + tn * 16 + (lane & 15)];
      }

#pragma unroll
    for (int tm = 0; tm < 2; ++tm)
#pragma unroll
      for (int tn = 0; tn < 4; ++tn)
        acc[tm][tn] = __builtin_amdgcn_wmma_f32_16x16x32_f16(
            false, afrag[tm], false, bfrag[tn], (short)0, acc[tm][tn],
            false, false);

    __syncthreads();  // release `buf` for the stage after next
  }

  // epilogue: C/D layout: VGPR v -> M = v + 8*(lane>>4), N = lane&15
  const int rb = (lane >> 4) * 8;
  const int cl = lane & 15;
#pragma unroll
  for (int tm = 0; tm < 2; ++tm) {
#pragma unroll
    for (int tn = 0; tn < 4; ++tn) {
      const int gcol = bn0 + wn * 64 + tn * 16 + cl;
#pragma unroll
      for (int v = 0; v < 8; ++v) {
        const int grow = bm0 + wm * 32 + tm * 16 + rb + v;
        if (grow < M) {
          float x = acc[tm][tn][v];
          if (bias) x += bias[gcol];
          if (posN) x += pos[(size_t)(grow % posN) * N + gcol];
          if (act == 1)      x = 0.5f * x * (1.0f + erff(x * 0.70710678f));
          else if (act == 2) x = fmaxf(x, 0.f);
          if (resid) x += (float)resid[(size_t)grow * N + gcol];
          if (outh) outh[(size_t)grow * N + gcol] = (_Float16)x;
          if (outf) outf[(size_t)grow * N + gcol] = x;
        }
      }
    }
  }
}

// ------------------------------------------------------------------
// Row LayerNorm, C=512, block per row. Optional f32 add before LN.
// ------------------------------------------------------------------
__global__ __launch_bounds__(256) void ln512_kernel(
    const _Float16* __restrict__ X, const float* __restrict__ addf,
    const float* __restrict__ g, const float* __restrict__ bt,
    _Float16* __restrict__ out) {
  const int row = blockIdx.x;
  const int tid = threadIdx.x;
  __shared__ float red[256];
  const size_t base = (size_t)row * C_DIM;
  float x0 = (float)X[base + tid];
  float x1 = (float)X[base + tid + 256];
  if (addf) { x0 += addf[base + tid]; x1 += addf[base + tid + 256]; }
  red[tid] = x0 + x1;
  __syncthreads();
  for (int o = 128; o > 0; o >>= 1) {
    if (tid < o) red[tid] += red[tid + o];
    __syncthreads();
  }
  const float mu = red[0] * (1.f / 512.f);
  __syncthreads();
  const float d0 = x0 - mu, d1 = x1 - mu;
  red[tid] = d0 * d0 + d1 * d1;
  __syncthreads();
  for (int o = 128; o > 0; o >>= 1) {
    if (tid < o) red[tid] += red[tid + o];
    __syncthreads();
  }
  const float rs = rsqrtf(red[0] * (1.f / 512.f) + 1e-5f);
  out[base + tid]       = (_Float16)(d0 * rs * g[tid] + bt[tid]);
  out[base + tid + 256] = (_Float16)(d1 * rs * g[tid + 256] + bt[tid + 256]);
}

// ------------------------------------------------------------------
// Landmark MLP: (B,136) -> relu @ (136,256) -> @ (256,512). Block per b.
// ------------------------------------------------------------------
__global__ __launch_bounds__(256) void lmk_mlp_kernel(
    const float* __restrict__ lm, const float* __restrict__ w1,
    const float* __restrict__ b1, const float* __restrict__ w2,
    const float* __restrict__ b2, _Float16* __restrict__ lmkh) {
  const int b = blockIdx.x;
  const int tid = threadIdx.x;
  __shared__ float lin[136];
  __shared__ float lf[256];
  if (tid < 136) lin[tid] = lm[b * 136 + tid];
  __syncthreads();
  {
    float a = b1[tid];
    for (int k = 0; k < 136; ++k) a += lin[k] * w1[k * 256 + tid];
    lf[tid] = fmaxf(a, 0.f);
  }
  __syncthreads();
  for (int c = tid; c < C_DIM; c += 256) {
    float a = b2[c];
    for (int k = 0; k < 256; ++k) a += lf[k] * w2[k * C_DIM + c];
    lmkh[(size_t)b * C_DIM + c] = (_Float16)a;
  }
}

// ------------------------------------------------------------------
// Single-query cross attention, block per (b,h). q f32 (B,C); K,V f16.
// ------------------------------------------------------------------
__global__ __launch_bounds__(256) void attn_lmk_kernel(
    const float* __restrict__ q, const _Float16* __restrict__ Km,
    const _Float16* __restrict__ Vm, _Float16* __restrict__ o, int n) {
  const int b = blockIdx.x >> 3;
  const int h = blockIdx.x & 7;
  const int tid = threadIdx.x;
  __shared__ float logits[1024];
  __shared__ float qs[HDIM];
  __shared__ float sred[256];
  __shared__ float oacc[256];
  const float sc = 0.125f;  // 64^-0.5

  if (tid < HDIM) qs[tid] = q[(size_t)b * C_DIM + h * HDIM + tid];
  __syncthreads();

  for (int i = tid; i < n; i += 256) {
    const _Float16* kr = Km + ((size_t)b * n + i) * C_DIM + h * HDIM;
    float s = 0.f;
    for (int d = 0; d < HDIM; ++d) s += qs[d] * (float)kr[d];
    logits[i] = s * sc;
  }
  __syncthreads();

  float m = -1e30f;
  for (int i = tid; i < n; i += 256) m = fmaxf(m, logits[i]);
  sred[tid] = m;
  __syncthreads();
  for (int s = 128; s > 0; s >>= 1) {
    if (tid < s) sred[tid] = fmaxf(sred[tid], sred[tid + s]);
    __syncthreads();
  }
  m = sred[0];
  __syncthreads();

  float sum = 0.f;
  for (int i = tid; i < n; i += 256) {
    float e = expf(logits[i] - m);
    logits[i] = e;
    sum += e;
  }
  sred[tid] = sum;
  __syncthreads();
  for (int s = 128; s > 0; s >>= 1) {
    if (tid < s) sred[tid] += sred[tid + s];
    __syncthreads();
  }
  const float inv = 1.0f / sred[0];
  __syncthreads();

  const int d = tid & 63, ch = tid >> 6;  // 4 chunks
  float a = 0.f;
  for (int i = ch; i < n; i += 4)
    a += logits[i] * (float)Vm[((size_t)b * n + i) * C_DIM + h * HDIM + d];
  oacc[tid] = a;
  __syncthreads();
  if (tid < 64) {
    float r = oacc[tid] + oacc[tid + 64] + oacc[tid + 128] + oacc[tid + 192];
    o[(size_t)b * C_DIM + h * HDIM + tid] = (_Float16)(r * inv);
  }
}

// X[row,c] += o2[b,c] with b = row/n  (in-place f16)
__global__ void bcast_add_kernel(_Float16* __restrict__ X,
                                 const float* __restrict__ o2,
                                 int n, size_t total) {
  size_t i = (size_t)blockIdx.x * blockDim.x + threadIdx.x;
  if (i >= total) return;
  const int c = (int)(i & (C_DIM - 1));
  const size_t row = i >> 9;
  const int b = (int)(row / n);
  X[i] = (_Float16)((float)X[i] + o2[(size_t)b * C_DIM + c]);
}

// fused[(b*3+s)*512+c] = mean over n patches of X
__global__ void mean_patch_kernel(const _Float16* __restrict__ X,
                                  float* __restrict__ fused, int n, int s) {
  const int idx = blockIdx.x * blockDim.x + threadIdx.x;
  if (idx >= BATCH * C_DIM) return;
  const int b = idx >> 9, c = idx & (C_DIM - 1);
  float a = 0.f;
  for (int i = 0; i < n; ++i)
    a += (float)X[((size_t)b * n + i) * C_DIM + c];
  fused[((size_t)b * 3 + s) * C_DIM + c] = a / n;
}

// 3-token self-attention; qkv f32 (48,1536); one thread per (b,h)
__global__ void fusion_attn_kernel(const float* __restrict__ qkv,
                                   _Float16* __restrict__ o) {
  const int t = blockIdx.x * blockDim.x + threadIdx.x;
  if (t >= BATCH * NHEAD) return;
  const int b = t >> 3, h = t & 7;
  const float sc = 0.125f;
  float lg[3][3];
  for (int qi = 0; qi < 3; ++qi)
    for (int ki = 0; ki < 3; ++ki) {
      float s = 0.f;
      const float* qp = qkv + ((size_t)b * 3 + qi) * 1536 + h * HDIM;
      const float* kp = qkv + ((size_t)b * 3 + ki) * 1536 + 512 + h * HDIM;
      for (int d = 0; d < HDIM; ++d) s += qp[d] * kp[d];
      lg[qi][ki] = s * sc;
    }
  for (int qi = 0; qi < 3; ++qi) {
    float m = fmaxf(lg[qi][0], fmaxf(lg[qi][1], lg[qi][2]));
    float e0 = expf(lg[qi][0] - m), e1 = expf(lg[qi][1] - m), e2 = expf(lg[qi][2] - m);
    float inv = 1.f / (e0 + e1 + e2);
    for (int d = 0; d < HDIM; ++d) {
      float v0 = qkv[((size_t)b * 3 + 0) * 1536 + 1024 + h * HDIM + d];
      float v1 = qkv[((size_t)b * 3 + 1) * 1536 + 1024 + h * HDIM + d];
      float v2 = qkv[((size_t)b * 3 + 2) * 1536 + 1024 + h * HDIM + d];
      o[((size_t)b * 3 + qi) * C_DIM + h * HDIM + d] =
          (_Float16)((e0 * v0 + e1 * v1 + e2 * v2) * inv);
    }
  }
}

// out[b,c] = mean over 3 tokens (f32 output)
__global__ void mean3_kernel(const _Float16* __restrict__ X,
                             float* __restrict__ out) {
  const int idx = blockIdx.x * blockDim.x + threadIdx.x;
  if (idx >= BATCH * C_DIM) return;
  const int b = idx >> 9, c = idx & (C_DIM - 1);
  float a = (float)X[((size_t)b * 3 + 0) * C_DIM + c] +
            (float)X[((size_t)b * 3 + 1) * C_DIM + c] +
            (float)X[((size_t)b * 3 + 2) * C_DIM + c];
  out[idx] = a * (1.f / 3.f);
}

// ------------------------------------------------------------------
// Host orchestration
// ------------------------------------------------------------------
static inline void launch_gemm(hipStream_t st, const _Float16* A,
                               const _Float16* B, const float* bias,
                               const float* pos, int posN,
                               const _Float16* resid, _Float16* outh,
                               float* outf, int M, int N, int K, int act) {
  dim3 g((M + TILE_M - 1) / TILE_M, N / TILE_N);
  gemm_wmma_kernel<<<g, 256, 0, st>>>(A, B, bias, pos, resid, outh, outf,
                                      M, N, K, posN, act);
}

static inline void launch_cvt(hipStream_t st, const float* s, _Float16* d,
                              size_t n) {
  int grid = (int)((n + 255) / 256);
  if (grid > 16384) grid = 16384;
  cvt_f32_f16_kernel<<<grid, 256, 0, st>>>(s, d, n);
}

extern "C" void kernel_launch(void* const* d_in, const int* in_sizes, int n_in,
                              void* d_out, int out_size, void* d_ws,
                              size_t ws_size, hipStream_t stream) {
  if (n_in < 87) return;
  const float* images    = (const float*)d_in[0];
  const float* landmarks = (const float*)d_in[1];

  // fusion params: idx 2..13 (f0), 14..25 (f1); sorted key order
  struct Fus {
    const float *fb1, *fb2, *fw1, *fw2, *in_b, *in_w;
    const float *ln1b, *ln1w, *ln2b, *ln2w, *out_b, *out_w;
  } fus[2];
  for (int f = 0; f < 2; ++f) {
    int i = 2 + f * 12;
    fus[f].fb1  = (const float*)d_in[i + 0];
    fus[f].fb2  = (const float*)d_in[i + 1];
    fus[f].fw1  = (const float*)d_in[i + 2];
    fus[f].fw2  = (const float*)d_in[i + 3];
    fus[f].in_b = (const float*)d_in[i + 4];
    fus[f].in_w = (const float*)d_in[i + 5];
    fus[f].ln1b = (const float*)d_in[i + 6];
    fus[f].ln1w = (const float*)d_in[i + 7];
    fus[f].ln2b = (const float*)d_in[i + 8];
    fus[f].ln2w = (const float*)d_in[i + 9];
    fus[f].out_b = (const float*)d_in[i + 10];
    fus[f].out_w = (const float*)d_in[i + 11];
  }
  const float* lmk_b1 = (const float*)d_in[26];
  const float* lmk_b2 = (const float*)d_in[27];
  const float* lmk_w1 = (const float*)d_in[28];
  const float* lmk_w2 = (const float*)d_in[29];

  struct Scale {
    const float *b1, *b2, *kb, *kw, *n1b, *n1w, *n2b, *n2w;
    const float *ob, *ow, *qb, *qw, *vb, *vw, *w1, *w2;
    const float *patch_b, *patch_w, *pos;
  } sp[3];
  for (int s = 0; s < 3; ++s) {
    int i = 30 + s * 19;
    sp[s].b1 = (const float*)d_in[i + 0];
    sp[s].b2 = (const float*)d_in[i + 1];
    sp[s].kb = (const float*)d_in[i + 2];
    sp[s].kw = (const float*)d_in[i + 3];
    sp[s].n1b = (const float*)d_in[i + 4];
    sp[s].n1w = (const float*)d_in[i + 5];
    sp[s].n2b = (const float*)d_in[i + 6];
    sp[s].n2w = (const float*)d_in[i + 7];
    sp[s].ob = (const float*)d_in[i + 8];
    sp[s].ow = (const float*)d_in[i + 9];
    sp[s].qb = (const float*)d_in[i + 10];
    sp[s].qw = (const float*)d_in[i + 11];
    sp[s].vb = (const float*)d_in[i + 12];
    sp[s].vw = (const float*)d_in[i + 13];
    sp[s].w1 = (const float*)d_in[i + 14];
    sp[s].w2 = (const float*)d_in[i + 15];
    sp[s].patch_b = (const float*)d_in[i + 16];
    sp[s].patch_w = (const float*)d_in[i + 17];
    sp[s].pos     = (const float*)d_in[i + 18];
  }

  const int psz[3]   = {28, 14, 7};
  const int nside[3] = {8, 16, 32};
  const int npat[3]  = {64, 256, 1024};
  int kreal[3], kpad[3];
  for (int s = 0; s < 3; ++s) {
    kreal[s] = 3 * psz[s] * psz[s];
    kpad[s]  = (kreal[s] + 31) & ~31;
  }

  // workspace bump allocator
  char* wp = (char*)d_ws;
  auto alloc = [&](size_t bytes) -> void* {
    void* r = (void*)wp;
    wp += (bytes + 255) & ~(size_t)255;
    return r;
  };
  const size_t CC  = (size_t)C_DIM * C_DIM;      // 262144
  const size_t CM  = (size_t)C_DIM * MLP_DIM;    // 1048576

  _Float16 *Wk[3], *Wq[3], *Wv[3], *Wo[3], *W1[3], *W2[3], *Wp[3];
  for (int s = 0; s < 3; ++s) {
    Wk[s] = (_Float16*)alloc(DEPTH_N * CC * 2);
    Wq[s] = (_Float16*)alloc(DEPTH_N * CC * 2);
    Wv[s] = (_Float16*)alloc(DEPTH_N * CC * 2);
    Wo[s] = (_Float16*)alloc(DEPTH_N * CC * 2);
    W1[s] = (_Float16*)alloc(DEPTH_N * CM * 2);
    W2[s] = (_Float16*)alloc(DEPTH_N * CM * 2);
    Wp[s] = (_Float16*)alloc((size_t)kpad[s] * C_DIM * 2);
  }
  _Float16 *Win[2], *Wout[2], *Wf1[2], *Wf2[2];
  for (int f = 0; f < 2; ++f) {
    Win[f]  = (_Float16*)alloc((size_t)C_DIM * 3 * C_DIM * 2);
    Wout[f] = (_Float16*)alloc(CC * 2);
    Wf1[f]  = (_Float16*)alloc(CM * 2);
    Wf2[f]  = (_Float16*)alloc(CM * 2);
  }

  const size_t MMAX = (size_t)BATCH * 1024;  // 16384 rows max
  _Float16* Xp  = (_Float16*)alloc(MMAX * 192 * 2);        // >= max M*Kpad
  _Float16* X   = (_Float16*)alloc(MMAX * C_DIM * 2);
  _Float16* LNX = (_Float16*)alloc(MMAX * C_DIM * 2);
  _Float16* Kb  = (_Float16*)alloc(MMAX * C_DIM * 2);
  _Float16* Vb  = (_Float16*)alloc(MMAX * C_DIM * 2);
  _Float16* HID = (_Float16*)alloc(MMAX * MLP_DIM * 2);
  _Float16* lmkh = (_Float16*)alloc((size_t)BATCH * C_DIM * 2);
  _Float16* lnlh = (_Float16*)alloc((size_t)BATCH * C_DIM * 2);
  float*    qf   = (float*)alloc((size_t)BATCH * C_DIM * 4);
  _Float16* oh   = (_Float16*)alloc((size_t)BATCH * C_DIM * 2);
  float*    o2f  = (float*)alloc((size_t)BATCH * C_DIM * 4);
  float*    fused = (float*)alloc((size_t)BATCH * 3 * C_DIM * 4);
  _Float16* Xf    = (_Float16*)alloc((size_t)BATCH * 3 * C_DIM * 2);
  float*    qkvf  = (float*)alloc((size_t)BATCH * 3 * 1536 * 4);
  _Float16* ah    = (_Float16*)alloc((size_t)BATCH * 3 * C_DIM * 2);
  float*    opf   = (float*)alloc((size_t)BATCH * 3 * C_DIM * 4);
  _Float16* fhid  = (_Float16*)alloc((size_t)BATCH * 3 * MLP_DIM * 2);
  float*    ffo   = (float*)alloc((size_t)BATCH * 3 * C_DIM * 4);

  // ---- weight conversion (f32 -> f16), once per launch ----
  for (int s = 0; s < 3; ++s) {
    launch_cvt(stream, sp[s].kw, Wk[s], DEPTH_N * CC);
    launch_cvt(stream, sp[s].qw, Wq[s], DEPTH_N * CC);
    launch_cvt(stream, sp[s].vw, Wv[s], DEPTH_N * CC);
    launch_cvt(stream, sp[s].ow, Wo[s], DEPTH_N * CC);
    launch_cvt(stream, sp[s].w1, W1[s], DEPTH_N * CM);
    launch_cvt(stream, sp[s].w2, W2[s], DEPTH_N * CM);
    size_t tot = (size_t)kpad[s] * C_DIM;
    int grid = (int)((tot + 255) / 256); if (grid > 16384) grid = 16384;
    cvt_pad_rows_kernel<<<grid, 256, 0, stream>>>(sp[s].patch_w, Wp[s],
                                                  kreal[s], kpad[s], C_DIM);
  }
  for (int f = 0; f < 2; ++f) {
    launch_cvt(stream, fus[f].in_w, Win[f], (size_t)C_DIM * 3 * C_DIM);
    launch_cvt(stream, fus[f].out_w, Wout[f], CC);
    launch_cvt(stream, fus[f].fw1, Wf1[f], CM);
    launch_cvt(stream, fus[f].fw2, Wf2[f], CM);
  }

  // ---- landmark token ----
  lmk_mlp_kernel<<<BATCH, 256, 0, stream>>>(landmarks, lmk_w1, lmk_b1,
                                            lmk_w2, lmk_b2, lmkh);

  // ---- per-scale transformer blocks ----
  for (int s = 0; s < 3; ++s) {
    const int n = npat[s];
    const int M = BATCH * n;
    const size_t total = (size_t)M * C_DIM;

    {
      size_t tp = (size_t)M * kpad[s];
      int grid = (int)((tp + 255) / 256); if (grid > 16384) grid = 16384;
      patchify_kernel<<<grid, 256, 0, stream>>>(images, Xp, psz[s], nside[s],
                                                kreal[s], kpad[s]);
    }
    // X = Xp @ Wp + patch_b + pos
    launch_gemm(stream, Xp, Wp[s], sp[s].patch_b, sp[s].pos, n, nullptr,
                X, nullptr, M, C_DIM, kpad[s], 0);

    for (int d = 0; d < DEPTH_N; ++d) {
      const float* n1w = sp[s].n1w + d * C_DIM;
      const float* n1b = sp[s].n1b + d * C_DIM;
      const float* n2w = sp[s].n2w + d * C_DIM;
      const float* n2b = sp[s].n2b + d * C_DIM;

      ln512_kernel<<<BATCH, 256, 0, stream>>>(lmkh, nullptr, n1w, n1b, lnlh);
      ln512_kernel<<<M, 256, 0, stream>>>(X, nullptr, n1w, n1b, LNX);

      launch_gemm(stream, lnlh, Wq[s] + d * CC, sp[s].qb + d * C_DIM,
                  nullptr, 0, nullptr, nullptr, qf, BATCH, C_DIM, C_DIM, 0);
      launch_gemm(stream, LNX, Wk[s] + d * CC, sp[s].kb + d * C_DIM,
                  nullptr, 0, nullptr, Kb, nullptr, M, C_DIM, C_DIM, 0);
      launch_gemm(stream, LNX, Wv[s] + d * CC, sp[s].vb + d * C_DIM,
                  nullptr, 0, nullptr, Vb, nullptr, M, C_DIM, C_DIM, 0);

      attn_lmk_kernel<<<BATCH * NHEAD, 256, 0, stream>>>(qf, Kb, Vb, oh, n);

      launch_gemm(stream, oh, Wo[s] + d * CC, sp[s].ob + d * C_DIM,
                  nullptr, 0, nullptr, nullptr, o2f, BATCH, C_DIM, C_DIM, 0);

      {
        int grid = (int)((total + 255) / 256);
        bcast_add_kernel<<<grid, 256, 0, stream>>>(X, o2f, n, total);
      }

      ln512_kernel<<<M, 256, 0, stream>>>(X, nullptr, n2w, n2b, LNX);
      launch_gemm(stream, LNX, W1[s] + d * CM, sp[s].b1 + d * MLP_DIM,
                  nullptr, 0, nullptr, HID, nullptr, M, MLP_DIM, C_DIM, 1);
      launch_gemm(stream, HID, W2[s] + d * CM, sp[s].b2 + d * C_DIM,
                  nullptr, 0, X, X, nullptr, M, C_DIM, MLP_DIM, 0);
    }

    mean_patch_kernel<<<(BATCH * C_DIM + 255) / 256, 256, 0, stream>>>(
        X, fused, n, s);
  }

  // ---- fusion blocks over (B,3,C) ----
  launch_cvt(stream, fused, Xf, (size_t)BATCH * 3 * C_DIM);
  const int MF = BATCH * 3;  // 48 rows
  for (int f = 0; f < 2; ++f) {
    launch_gemm(stream, Xf, Win[f], fus[f].in_b, nullptr, 0, nullptr,
                nullptr, qkvf, MF, 3 * C_DIM, C_DIM, 0);
    fusion_attn_kernel<<<1, 128, 0, stream>>>(qkvf, ah);
    launch_gemm(stream, ah, Wout[f], fus[f].out_b, nullptr, 0, nullptr,
                nullptr, opf, MF, C_DIM, C_DIM, 0);
    ln512_kernel<<<MF, 256, 0, stream>>>(Xf, opf, fus[f].ln1w, fus[f].ln1b, Xf);
    launch_gemm(stream, Xf, Wf1[f], fus[f].fb1, nullptr, 0, nullptr,
                fhid, nullptr, MF, MLP_DIM, C_DIM, 2);
    launch_gemm(stream, fhid, Wf2[f], fus[f].fb2, nullptr, 0, nullptr,
                nullptr, ffo, MF, C_DIM, MLP_DIM, 0);
    ln512_kernel<<<MF, 256, 0, stream>>>(Xf, ffo, fus[f].ln2w, fus[f].ln2b, Xf);
  }

  mean3_kernel<<<(BATCH * C_DIM + 255) / 256, 256, 0, stream>>>(
      Xf, (float*)d_out);
}
